// NodeModel_6030134084155
// MI455X (gfx1250) — compile-verified, matrix operand order checked by reading, not run on hardware
//
#include <hip/hip_runtime.h>
#include <hip/hip_bf16.h>

typedef __attribute__((ext_vector_type(16))) _Float16     v16h;
typedef __attribute__((ext_vector_type(8)))  float        v8f;
typedef __attribute__((ext_vector_type(4)))  unsigned int u32x4;

#define DD    64     // feature dim
#define KDIM  128    // concat dim
#define RSTR  136    // padded LDS row stride in halves (272B, 16B aligned, bank-spread)
#define EPSV  1e-5f

union F16Frag { v16h v; u32x4 q[2]; };
struct __align__(8) H4 { _Float16 a, b, c, d; };

__device__ __forceinline__ float silu_f(float v) {
    return v * (1.0f / (1.0f + __expf(-v)));
}

// ---------------- zero init ----------------
__global__ void zero_kernel(float* __restrict__ p, long long n) {
    long long i = (long long)blockIdx.x * blockDim.x + threadIdx.x;
    if (i < n) p[i] = 0.0f;
}

// ---------------- BN finalize: a = g*rsqrt(var+eps), sh = be - mu*a ----------------
__global__ void bn_finalize_kernel(const float* __restrict__ sum, const float* __restrict__ sq,
                                   const float* __restrict__ g,   const float* __restrict__ be,
                                   float invM, float* __restrict__ a, float* __restrict__ sh) {
    int n = threadIdx.x;
    if (n < DD) {
        float mu  = sum[n] * invM;
        float var = fmaxf(sq[n] * invM - mu * mu, 0.0f);
        float r   = rsqrtf(var + EPSV);
        float an  = g[n] * r;
        a[n]  = an;
        sh[n] = be[n] - mu * an;
    }
}

// ---------------- Edge GEMM: concat(x[row], edge_attr) @ W1 ----------------
// MODE 0: accumulate column sum / sumsq for BN (no stores of h1).
// MODE 1: recompute GEMM, apply BN affine + SiLU, atomic scatter into sums/cnt.
template<int MODE>
__global__ __launch_bounds__(256) void edge_gemm_kernel(
    const float* __restrict__ x, const float* __restrict__ ea,
    const int* __restrict__ row, const int* __restrict__ col,
    const float* __restrict__ W1,
    float* __restrict__ stSum, float* __restrict__ stSq,          // MODE 0
    const float* __restrict__ A1, const float* __restrict__ SH1,  // MODE 1
    float* __restrict__ sums, float* __restrict__ cnt,            // MODE 1
    int E)
{
    __shared__ __align__(16) _Float16 ldsW[DD * RSTR];      // W^T: [n][k]
    __shared__ __align__(16) _Float16 ldsA[8 * 16 * RSTR];  // per-wave 16x128 A tiles
    __shared__ float bS[DD], bQ[DD];

    const int tid  = threadIdx.x;
    const int wave = tid >> 5;
    const int lane = tid & 31;
    const int l16  = lane & 15;
    const int hi   = lane >> 4;

    if (MODE == 0 && tid < DD) { bS[tid] = 0.0f; bQ[tid] = 0.0f; }

    // stage W transposed as f16 (contiguous LDS writes)
    for (int i = tid; i < DD * KDIM; i += 256) {
        int n = i >> 7, k = i & 127;
        ldsW[n * RSTR + k] = (_Float16)W1[(size_t)k * DD + n];
    }

    // stage A: 16 edges per wave, concat(x[row[e]], edge_attr[e]) -> f16
    const int e0 = blockIdx.x * 128 + wave * 16;
    _Float16* aw = ldsA + wave * (16 * RSTR);
    {
        const int k = lane * 4;  // 0..124
        for (int j = 0; j < 16; ++j) {
            int e  = e0 + j;
            int ec = (e < E) ? e : 0;
            float4 f;
            if (k < DD) {
                int r = row[ec];
                f = *(const float4*)(x + (size_t)r * DD + k);
            } else {
                f = *(const float4*)(ea + (size_t)ec * DD + (k - DD));
            }
            H4 p = {(_Float16)f.x, (_Float16)f.y, (_Float16)f.z, (_Float16)f.w};
            *(H4*)(aw + j * RSTR + k) = p;
        }
    }
    __syncthreads();

    // 16x64 C tile per wave: 4 accumulators, 4 K-steps of v_wmma_f32_16x16x32_f16
    v8f acc[4] = {};
    const int kb = hi * 8;
#pragma unroll
    for (int kk = 0; kk < 4; ++kk) {
        F16Frag A;
        const _Float16* ap = aw + l16 * RSTR + kk * 32 + kb;
        A.q[0] = *(const u32x4*)ap;
        A.q[1] = *(const u32x4*)(ap + 16);
#pragma unroll
        for (int nt = 0; nt < 4; ++nt) {
            F16Frag B;
            const _Float16* bp = ldsW + (nt * 16 + l16) * RSTR + kk * 32 + kb;
            B.q[0] = *(const u32x4*)bp;
            B.q[1] = *(const u32x4*)(bp + 16);
            acc[nt] = __builtin_amdgcn_wmma_f32_16x16x32_f16(
                false, A.v, false, B.v, (short)0, acc[nt], false, false);
        }
    }

    if (MODE == 0) {
#pragma unroll
        for (int nt = 0; nt < 4; ++nt) {
            int n = nt * 16 + l16;
            float s = 0.0f, q = 0.0f;
#pragma unroll
            for (int r = 0; r < 8; ++r) {
                int e = e0 + hi * 8 + r;
                if (e < E) { float v = acc[nt][r]; s += v; q += v * v; }
            }
            atomicAdd(&bS[n], s);
            atomicAdd(&bQ[n], q);
        }
        __syncthreads();
        if (tid < DD) { atomicAdd(&stSum[tid], bS[tid]); atomicAdd(&stSq[tid], bQ[tid]); }
    } else {
        int c8[8];
#pragma unroll
        for (int r = 0; r < 8; ++r) {
            int e = e0 + hi * 8 + r;
            c8[r] = col[(e < E) ? e : 0];
        }
#pragma unroll
        for (int nt = 0; nt < 4; ++nt) {
            int n = nt * 16 + l16;
            float an = A1[n], shn = SH1[n];
#pragma unroll
            for (int r = 0; r < 8; ++r) {
                int e = e0 + hi * 8 + r;
                if (e < E) {
                    float v = acc[nt][r] * an + shn;
                    atomicAdd(&sums[(size_t)c8[r] * DD + n], silu_f(v));
                }
            }
        }
        if (hi == 0) {
            int e = e0 + l16;
            if (e < E) atomicAdd(&cnt[col[e]], 1.0f);
        }
    }
}

// ---------------- Node GEMM: concat(x, sums/max(cnt,1)) @ W2, store h2 + BN stats ----------------
__global__ __launch_bounds__(256) void node_gemm_kernel(
    const float* __restrict__ x, const float* __restrict__ sums, const float* __restrict__ cnt,
    const float* __restrict__ W2,
    float* __restrict__ h2, float* __restrict__ stSum, float* __restrict__ stSq, int N)
{
    __shared__ __align__(16) _Float16 ldsW[DD * RSTR];
    __shared__ __align__(16) _Float16 ldsA[8 * 16 * RSTR];
    __shared__ float bS[DD], bQ[DD];

    const int tid  = threadIdx.x;
    const int wave = tid >> 5;
    const int lane = tid & 31;
    const int l16  = lane & 15;
    const int hi   = lane >> 4;

    if (tid < DD) { bS[tid] = 0.0f; bQ[tid] = 0.0f; }

    for (int i = tid; i < DD * KDIM; i += 256) {
        int n = i >> 7, k = i & 127;
        ldsW[n * RSTR + k] = (_Float16)W2[(size_t)k * DD + n];
    }

    const int i0 = blockIdx.x * 128 + wave * 16;
    _Float16* aw = ldsA + wave * (16 * RSTR);
    {
        const int k = lane * 4;
        for (int j = 0; j < 16; ++j) {
            int i  = i0 + j;
            int ic = (i < N) ? i : 0;
            float4 f;
            if (k < DD) {
                f = *(const float4*)(x + (size_t)ic * DD + k);
            } else {
                float inv = 1.0f / fmaxf(cnt[ic], 1.0f);
                float4 s4 = *(const float4*)(sums + (size_t)ic * DD + (k - DD));
                f.x = s4.x * inv; f.y = s4.y * inv; f.z = s4.z * inv; f.w = s4.w * inv;
            }
            H4 p = {(_Float16)f.x, (_Float16)f.y, (_Float16)f.z, (_Float16)f.w};
            *(H4*)(aw + j * RSTR + k) = p;
        }
    }
    __syncthreads();

    v8f acc[4] = {};
    const int kb = hi * 8;
#pragma unroll
    for (int kk = 0; kk < 4; ++kk) {
        F16Frag A;
        const _Float16* ap = aw + l16 * RSTR + kk * 32 + kb;
        A.q[0] = *(const u32x4*)ap;
        A.q[1] = *(const u32x4*)(ap + 16);
#pragma unroll
        for (int nt = 0; nt < 4; ++nt) {
            F16Frag B;
            const _Float16* bp = ldsW + (nt * 16 + l16) * RSTR + kk * 32 + kb;
            B.q[0] = *(const u32x4*)bp;
            B.q[1] = *(const u32x4*)(bp + 16);
            acc[nt] = __builtin_amdgcn_wmma_f32_16x16x32_f16(
                false, A.v, false, B.v, (short)0, acc[nt], false, false);
        }
    }

#pragma unroll
    for (int nt = 0; nt < 4; ++nt) {
        int n = nt * 16 + l16;
        float s = 0.0f, q = 0.0f;
#pragma unroll
        for (int r = 0; r < 8; ++r) {
            int i = i0 + hi * 8 + r;
            if (i < N) {
                float v = acc[nt][r];
                s += v; q += v * v;
                h2[(size_t)i * DD + n] = v;
            }
        }
        atomicAdd(&bS[n], s);
        atomicAdd(&bQ[n], q);
    }
    __syncthreads();
    if (tid < DD) { atomicAdd(&stSum[tid], bS[tid]); atomicAdd(&stSq[tid], bQ[tid]); }
}

// ---------------- Final elementwise: out = SiLU(h2*a + sh) ----------------
__global__ void node_out_kernel(const float* __restrict__ h2, const float* __restrict__ a,
                                const float* __restrict__ sh, float* __restrict__ out, int n4) {
    int i = blockIdx.x * blockDim.x + threadIdx.x;
    if (i < n4) {
        float4 v = ((const float4*)h2)[i];
        int n = (i & 15) * 4;
        float4 o;
        o.x = silu_f(v.x * a[n]     + sh[n]);
        o.y = silu_f(v.y * a[n + 1] + sh[n + 1]);
        o.z = silu_f(v.z * a[n + 2] + sh[n + 2]);
        o.w = silu_f(v.w * a[n + 3] + sh[n + 3]);
        ((float4*)out)[i] = o;
    }
}

extern "C" void kernel_launch(void* const* d_in, const int* in_sizes, int n_in,
                              void* d_out, int out_size, void* d_ws, size_t ws_size,
                              hipStream_t stream) {
    const float* x   = (const float*)d_in[0];
    const float* ea  = (const float*)d_in[1];
    // d_in[2] = u (unused by reference)
    const float* W1  = (const float*)d_in[3];
    // d_in[4] = b1 : cancelled by BatchNorm mean subtraction
    const float* g1  = (const float*)d_in[5];
    const float* be1 = (const float*)d_in[6];
    const float* W2  = (const float*)d_in[7];
    // d_in[8] = b2 : cancelled by BatchNorm mean subtraction
    const float* g2  = (const float*)d_in[9];
    const float* be2 = (const float*)d_in[10];
    const int*   ei  = (const int*)d_in[11];
    // d_in[12] = batch (unused by reference)

    const int N = in_sizes[0] / DD;   // 100000
    const int E = in_sizes[1] / DD;   // 1000000
    const int* row = ei;
    const int* col = ei + E;

    float* ws   = (float*)d_ws;
    float* sums = ws;                          // N*64
    float* cnt  = sums + (size_t)N * DD;       // N
    float* st   = cnt + N;                     // 512 floats of stats
    float* sum1 = st;        float* sq1 = st + 64;
    float* a1   = st + 128;  float* sh1 = st + 192;
    float* sum2 = st + 256;  float* sq2 = st + 320;
    float* a2   = st + 384;  float* sh2 = st + 448;
    float* h2   = st + 512;                    // N*64

    // 1) zero sums/cnt/stats
    long long nz = (long long)N * DD + N + 512;
    zero_kernel<<<(int)((nz + 255) / 256), 256, 0, stream>>>(ws, nz);

    int eblocks = (E + 127) / 128;
    // 2) edge GEMM pass 1: BN batch stats
    edge_gemm_kernel<0><<<eblocks, 256, 0, stream>>>(
        x, ea, row, col, W1, sum1, sq1, nullptr, nullptr, nullptr, nullptr, E);
    // 3) finalize BN1
    bn_finalize_kernel<<<1, 64, 0, stream>>>(sum1, sq1, g1, be1, 1.0f / (float)E, a1, sh1);
    // 4) edge GEMM pass 2: norm + SiLU + scatter_mean accumulation
    edge_gemm_kernel<1><<<eblocks, 256, 0, stream>>>(
        x, ea, row, col, W1, nullptr, nullptr, a1, sh1, sums, cnt, E);

    // 5) node GEMM + BN2 stats
    int nblocks = (N + 127) / 128;
    node_gemm_kernel<<<nblocks, 256, 0, stream>>>(x, sums, cnt, W2, h2, sum2, sq2, N);
    // 6) finalize BN2
    bn_finalize_kernel<<<1, 64, 0, stream>>>(sum2, sq2, g2, be2, 1.0f / (float)N, a2, sh2);
    // 7) elementwise output
    node_out_kernel<<<(N * 16 + 255) / 256, 256, 0, stream>>>(h2, a2, sh2, (float*)d_out, N * 16);

    (void)n_in; (void)out_size; (void)ws_size;
}
